// RegionProposalNetwork_893353198233
// MI455X (gfx1250) — matrix operand order repaired and checked by reading, not compiled, not run on hardware
//
#include <hip/hip_runtime.h>
#include <hip/hip_bf16.h>

typedef __attribute__((ext_vector_type(16))) _Float16 v16h;
typedef __attribute__((ext_vector_type(8)))  _Float16 v8h;
typedef __attribute__((ext_vector_type(8)))  float    v8f;

#define C       256
#define POS     4096      // 64*64 spatial positions
#define K1      2304      // 9*256
#define NH      48        // 9 cls + 36 reg + 3 pad
#define NANCH   36864     // 4096*9
#define SORTN   65536
#define TOPK    4000
#define SORT2N  4096
#define OUTK    2000
#define DW_CLAMP 4.135166556742356f

union Frag { v16h v; v8h h[2]; };

__device__ __forceinline__ unsigned long long makeKey(float s, unsigned idx) {
  unsigned b = __float_as_uint(s);
  b = (b & 0x80000000u) ? ~b : (b | 0x80000000u);   // order-preserving float->uint
  return ((unsigned long long)b << 32) | (unsigned)(0xFFFFFFFFu - idx);
}

// Async copy of 16 bytes global -> LDS, tracked by ASYNCcnt (CDNA5 path).
__device__ __forceinline__ void async_b128(unsigned lds_off, const void* gaddr) {
  asm volatile("global_load_async_to_lds_b128 %0, %1, off"
               :: "v"(lds_off), "v"((unsigned long long)gaddr)
               : "memory");
}
// Wait until at most N async ops remain outstanding (in-order completion per wave).
__device__ __forceinline__ void wait_async_le5() {
  asm volatile("s_wait_asynccnt 0x5" ::: "memory");
}
__device__ __forceinline__ void wait_async0() {
  asm volatile("s_wait_asynccnt 0x0" ::: "memory");
}

// ---------------- im2col: feat (256,64,64) f32 -> A [4096][2304] f16 ----------------
__global__ void k_im2col(const float* __restrict__ feat, _Float16* __restrict__ A) {
  int idx = blockIdx.x * blockDim.x + threadIdx.x;
  if (idx >= POS * K1) return;
  int p = idx / K1, k = idx - p * K1;
  int tap = k >> 8, ic = k & 255;
  int ky = tap / 3, kx = tap - ky * 3;
  int h = p >> 6, w = p & 63;
  int ih = h + ky - 1, iw = w + kx - 1;
  float v = 0.f;
  if ((unsigned)ih < 64u && (unsigned)iw < 64u) v = feat[ic * 4096 + ih * 64 + iw];
  A[idx] = (_Float16)v;
}

// ---------------- pack conv1_w (256,256,3,3) -> B1 [oc][k] f16 ----------------
__global__ void k_pack_b1(const float* __restrict__ w, _Float16* __restrict__ B) {
  int idx = blockIdx.x * blockDim.x + threadIdx.x;
  if (idx >= C * K1) return;
  int oc = idx / K1, k = idx - oc * K1;
  int tap = k >> 8, ic = k & 255;
  B[idx] = (_Float16)w[(oc * 256 + ic) * 9 + tap];
}

// ---------------- pack cls_w (9,256) + reg_w (36,256) -> B2 [48][256] f16 + bias2[48] ----------------
__global__ void k_pack_b2(const float* __restrict__ clsw, const float* __restrict__ regw,
                          const float* __restrict__ clsb, const float* __restrict__ regb,
                          _Float16* __restrict__ B, float* __restrict__ bias) {
  int idx = blockIdx.x * blockDim.x + threadIdx.x;
  if (idx < NH) bias[idx] = (idx < 9) ? clsb[idx] : (idx < 45 ? regb[idx - 9] : 0.f);
  if (idx >= NH * C) return;
  int n = idx / C, ic = idx - n * C;
  float v = 0.f;
  if (n < 9) v = clsw[n * C + ic];
  else if (n < 45) v = regw[(n - 9) * C + ic];
  B[idx] = (_Float16)v;
}

// ---------------- GEMM1: X[4096][256] = relu(A[4096][2304] * B1^T + conv1_b), f16 out ----------------
// Double-buffered async staging: issue panel k+1 with GLOBAL_LOAD_ASYNC_TO_LDS_B128, then
// s_wait_asynccnt 0x5 so panel k (issued last iteration) is resident while k+1 streams in.
// Each wave computes a 32x32 output tile: 4 WMMAs per 32-deep K panel.
__global__ __launch_bounds__(128) void k_gemm1(const _Float16* __restrict__ A,
                                               const _Float16* __restrict__ B,
                                               const float* __restrict__ bias,
                                               _Float16* __restrict__ X) {
  __shared__ __align__(16) _Float16 sA[2 * 128 * 32];   // 2 x (128 rows x 64B)
  __shared__ __align__(16) _Float16 sB[2 * 32 * 32];    // 2 x (32 rows x 64B)
  const int m0 = blockIdx.x * 128, n0 = blockIdx.y * 32;
  const int t = threadIdx.x, lane = t & 31, wv = t >> 5;
  v8f acc00 = {}, acc01 = {}, acc10 = {}, acc11 = {};
  const int arow0 = wv * 32 + (lane & 15);
  const int ah = (lane & 16) ? 16 : 0;         // A fragment K sub-base
  const int bk = (lane & 16) ? 16 : 0;         // B fragment K sub-base
  const int bcol = lane & 15;

  // Async staging: 16B chunks. A tile 128x64B = 512 chunks (4/thread), B tile 32x64B = 128 (1/thread).
  const unsigned ldsA = (unsigned)(size_t)(void*)sA;
  const unsigned ldsB = (unsigned)(size_t)(void*)sB;
  const int sub = t & 3;
  const char* gA[4];
  #pragma unroll
  for (int q = 0; q < 4; ++q)
    gA[q] = (const char*)(A + (size_t)(m0 + (t >> 2) + 32 * q) * K1) + sub * 16;
  const char* gB0 = (const char*)(B + (size_t)(n0 + (t >> 2)) * K1) + sub * 16;

  #define ISSUE_PANEL(buf, kk)                                               \
    do {                                                                     \
      const size_t koff_ = (size_t)(kk) * 2;                                 \
      const unsigned aoff_ = ldsA + (unsigned)(buf) * 8192u;                 \
      _Pragma("unroll")                                                      \
      for (int q_ = 0; q_ < 4; ++q_)                                         \
        async_b128(aoff_ + (unsigned)(t + 128 * q_) * 16u, gA[q_] + koff_);  \
      async_b128(ldsB + (unsigned)(buf) * 2048u + (unsigned)t * 16u,         \
                 gB0 + koff_);                                               \
    } while (0)

  ISSUE_PANEL(0, 0);
  for (int kk = 0; kk < K1; kk += 32) {
    const int cur = (kk >> 5) & 1;
    if (kk + 32 < K1) { ISSUE_PANEL(cur ^ 1, kk + 32); wait_async_le5(); }
    else              { wait_async0(); }
    __syncthreads();
    const _Float16* ab0 = sA + cur * 4096 + arow0 * 32;
    const _Float16* bb  = sB + cur * 1024;
    Frag fa0, fa1;
    fa0.h[0] = *(const v8h*)(ab0 + ah);
    fa0.h[1] = *(const v8h*)(ab0 + ah + 8);
    fa1.h[0] = *(const v8h*)(ab0 + 16 * 32 + ah);
    fa1.h[1] = *(const v8h*)(ab0 + 16 * 32 + ah + 8);
    v16h b0 = *(const v16h*)(bb + bcol * 32 + bk);
    v16h b1 = *(const v16h*)(bb + (bcol + 16) * 32 + bk);
    acc00 = __builtin_amdgcn_wmma_f32_16x16x32_f16(false, fa0.v, false, b0, (short)0, acc00, false, false);
    acc01 = __builtin_amdgcn_wmma_f32_16x16x32_f16(false, fa0.v, false, b1, (short)0, acc01, false, false);
    acc10 = __builtin_amdgcn_wmma_f32_16x16x32_f16(false, fa1.v, false, b0, (short)0, acc10, false, false);
    acc11 = __builtin_amdgcn_wmma_f32_16x16x32_f16(false, fa1.v, false, b1, (short)0, acc11, false, false);
    __syncthreads();
  }
  #undef ISSUE_PANEL

  const int rbase0 = m0 + wv * 32 + ((lane >> 4) & 1) * 8;
  const int colL = lane & 15;
  for (int v = 0; v < 8; ++v) {
    int r0 = rbase0 + v, r1 = rbase0 + 16 + v;
    int c0 = n0 + colL, c1 = n0 + 16 + colL;
    float x00 = acc00[v] + bias[c0]; x00 = x00 < 0.f ? 0.f : x00;
    float x01 = acc01[v] + bias[c1]; x01 = x01 < 0.f ? 0.f : x01;
    float x10 = acc10[v] + bias[c0]; x10 = x10 < 0.f ? 0.f : x10;
    float x11 = acc11[v] + bias[c1]; x11 = x11 < 0.f ? 0.f : x11;
    X[(size_t)r0 * C + c0] = (_Float16)x00;
    X[(size_t)r0 * C + c1] = (_Float16)x01;
    X[(size_t)r1 * C + c0] = (_Float16)x10;
    X[(size_t)r1 * C + c1] = (_Float16)x11;
  }
}

// ---------------- GEMM2: head[4096][48] = X[4096][256] * B2^T + bias2, f32 out ----------------
__global__ __launch_bounds__(128) void k_gemm2(const _Float16* __restrict__ A,
                                               const _Float16* __restrict__ B,
                                               const float* __restrict__ bias,
                                               float* __restrict__ O) {
  __shared__ __align__(16) _Float16 sA[64 * 32];
  __shared__ __align__(16) _Float16 sB[48 * 32];
  const int m0 = blockIdx.x * 64;
  const int t = threadIdx.x, lane = t & 31, wv = t >> 5;
  v8f acc[3] = {};
  const int arow = wv * 16 + (lane & 15);
  const int ah = (lane & 16) ? 16 : 0;
  const int bk = (lane & 16) ? 16 : 0;
  const int bcol = lane & 15;
  for (int kk = 0; kk < C; kk += 32) {
    {
      int r = t >> 1, hsel = t & 1;
      const uint4* s = (const uint4*)(A + (size_t)(m0 + r) * C + kk) + hsel * 2;
      uint4* d = (uint4*)(sA + r * 32) + hsel * 2;
      d[0] = s[0]; d[1] = s[1];
      if (hsel == 0 && kk + 32 < C)
        __builtin_prefetch((const void*)(A + (size_t)(m0 + r) * C + kk + 32), 0, 1);
    }
    if (t < 96) {
      int n = t >> 1, hsel = t & 1;
      const uint4* s = (const uint4*)(B + (size_t)n * C + kk) + hsel * 2;
      uint4* d = (uint4*)(sB + n * 32) + hsel * 2;
      d[0] = s[0]; d[1] = s[1];
    }
    __syncthreads();
    Frag fa;
    const _Float16* ab = sA + arow * 32;
    fa.h[0] = *(const v8h*)(ab + ah);
    fa.h[1] = *(const v8h*)(ab + ah + 8);
    #pragma unroll
    for (int ns = 0; ns < 3; ++ns) {
      v16h bf = *(const v16h*)(sB + (ns * 16 + bcol) * 32 + bk);
      acc[ns] = __builtin_amdgcn_wmma_f32_16x16x32_f16(false, fa.v, false, bf, (short)0, acc[ns], false, false);
    }
    __syncthreads();
  }
  const int rbase = m0 + wv * 16 + ((lane >> 4) & 1) * 8;
  const int colL = lane & 15;
  #pragma unroll
  for (int ns = 0; ns < 3; ++ns)
    for (int v = 0; v < 8; ++v) {
      int grow = rbase + v, gc = ns * 16 + colL;
      O[(size_t)grow * NH + gc] = acc[ns][v] + bias[gc];
    }
}

// ---------------- proposals: decode anchors + deltas, sigmoid, clip, valid, sort keys ----------------
__global__ void k_props(const float* __restrict__ head, float* __restrict__ boxes,
                        float* __restrict__ eff, unsigned long long* __restrict__ keys) {
  int idx = blockIdx.x * blockDim.x + threadIdx.x;
  if (idx >= SORTN) return;
  if (idx >= NANCH) { keys[idx] = (unsigned)(0xFFFFFFFFu - idx); return; }
  int p = idx / 9, a = idx - p * 9;
  int h = p >> 6, w = p & 63;
  const float* hp = head + (size_t)p * NH;
  float score = 1.f / (1.f + __expf(-hp[a]));
  float dx = hp[9 + a * 4 + 0], dy = hp[9 + a * 4 + 1];
  float dw = fminf(hp[9 + a * 4 + 2], DW_CLAMP);
  float dh = fminf(hp[9 + a * 4 + 3], DW_CLAMP);
  int ri = a / 3, si = a - ri * 3;
  float hr = sqrtf((float)(ri + 1)), wr = 1.f / hr;
  float scale = (si == 0) ? 128.f : ((si == 1) ? 256.f : 512.f);
  float bx = rintf(wr * scale * 0.5f), by = rintf(hr * scale * 0.5f);
  float sx = w * 16.f, sy = h * 16.f;
  float x1 = sx - bx, y1 = sy - by, x2 = sx + bx, y2 = sy + by;
  float aw = x2 - x1, ahh = y2 - y1;
  float cx = x1 + 0.5f * aw, cy = y1 + 0.5f * ahh;
  float pcx = dx * aw + cx, pcy = dy * ahh + cy;
  float pw = __expf(dw) * aw, ph = __expf(dh) * ahh;
  float px1 = fminf(fmaxf(pcx - 0.5f * pw, 0.f), 1024.f);
  float py1 = fminf(fmaxf(pcy - 0.5f * ph, 0.f), 1024.f);
  float px2 = fminf(fmaxf(pcx + 0.5f * pw, 0.f), 1024.f);
  float py2 = fminf(fmaxf(pcy + 0.5f * ph, 0.f), 1024.f);
  bool valid = (px2 - px1 >= 16.f) && (py2 - py1 >= 16.f);
  boxes[idx * 4 + 0] = px1; boxes[idx * 4 + 1] = py1;
  boxes[idx * 4 + 2] = px2; boxes[idx * 4 + 3] = py2;
  eff[idx] = valid ? score : -1.f;
  keys[idx] = makeKey(score, (unsigned)idx);
}

// ---------------- global bitonic stage (descending, 65536 keys) ----------------
__global__ void k_bitonic(unsigned long long* __restrict__ keys, int k, int j) {
  int i = blockIdx.x * blockDim.x + threadIdx.x;
  int ixj = i ^ j;
  if (ixj > i) {
    unsigned long long a = keys[i], b = keys[ixj];
    bool sw = ((i & k) == 0) ? (a < b) : (a > b);
    if (sw) { keys[i] = b; keys[ixj] = a; }
  }
}

// ---------------- gather top-4000 ----------------
__global__ void k_gather(const unsigned long long* __restrict__ keys,
                         const float* __restrict__ boxesAll, const float* __restrict__ effAll,
                         float* __restrict__ selB, float* __restrict__ selE) {
  int s = blockIdx.x * blockDim.x + threadIdx.x;
  if (s >= SORT2N) return;
  if (s < TOPK) {
    unsigned idx = 0xFFFFFFFFu - (unsigned)keys[s];
    selB[s * 4 + 0] = boxesAll[idx * 4 + 0]; selB[s * 4 + 1] = boxesAll[idx * 4 + 1];
    selB[s * 4 + 2] = boxesAll[idx * 4 + 2]; selB[s * 4 + 3] = boxesAll[idx * 4 + 3];
    selE[s] = effAll[idx];
  } else {
    selB[s * 4 + 0] = 0.f; selB[s * 4 + 1] = 0.f; selB[s * 4 + 2] = 0.f; selB[s * 4 + 3] = 0.f;
    selE[s] = -3.f;
  }
}

// ---------------- in-LDS bitonic sort of 4096 (eff desc, slot asc) ----------------
__global__ __launch_bounds__(1024) void k_sort2(const float* __restrict__ selB, const float* __restrict__ selE,
                                                float* __restrict__ sB, float* __restrict__ sE) {
  __shared__ unsigned long long kk[SORT2N];
  int t = threadIdx.x;
  for (int i = t; i < SORT2N; i += 1024) kk[i] = makeKey(selE[i], (unsigned)i);
  __syncthreads();
  for (int k = 2; k <= SORT2N; k <<= 1)
    for (int j = k >> 1; j > 0; j >>= 1) {
      for (int i = t; i < SORT2N; i += 1024) {
        int ixj = i ^ j;
        if (ixj > i) {
          unsigned long long a = kk[i], b = kk[ixj];
          bool sw = ((i & k) == 0) ? (a < b) : (a > b);
          if (sw) { kk[i] = b; kk[ixj] = a; }
        }
      }
      __syncthreads();
    }
  for (int i = t; i < SORT2N; i += 1024) {
    unsigned s = 0xFFFFFFFFu - (unsigned)kk[i];
    sB[i * 4 + 0] = selB[s * 4 + 0]; sB[i * 4 + 1] = selB[s * 4 + 1];
    sB[i * 4 + 2] = selB[s * 4 + 2]; sB[i * 4 + 3] = selB[s * 4 + 3];
    sE[i] = selE[s];
  }
}

// ---------------- NMS: 4000 boxes in LDS, sequential-i parallel-j ----------------
__global__ __launch_bounds__(1024) void k_nms(const float* __restrict__ boxes, unsigned* __restrict__ keepOut) {
  __shared__ float4 sb[TOPK];
  __shared__ unsigned kw[(TOPK + 31) / 32];
  int t = threadIdx.x;
  for (int i = t; i < TOPK; i += 1024) sb[i] = ((const float4*)boxes)[i];
  for (int i = t; i < (TOPK + 31) / 32; i += 1024) kw[i] = 0xFFFFFFFFu;
  __syncthreads();
  for (int i = 0; i < TOPK; ++i) {
    if ((kw[i >> 5] >> (i & 31)) & 1u) {
      float4 bi = sb[i];
      float ai = (bi.z - bi.x) * (bi.w - bi.y);
      for (int j = i + 1 + t; j < TOPK; j += 1024) {
        float4 bj = sb[j];
        float xl = fmaxf(bi.x, bj.x), yt = fmaxf(bi.y, bj.y);
        float xr = fminf(bi.z, bj.z), yb = fminf(bi.w, bj.w);
        float inter = fmaxf(xr - xl, 0.f) * fmaxf(yb - yt, 0.f);
        float aj = (bj.z - bj.x) * (bj.w - bj.y);
        float iou = inter / (ai + aj - inter);   // 0/0 -> NaN -> no suppress (matches ref)
        if (iou > 0.7f) atomicAnd(&kw[j >> 5], ~(1u << (j & 31)));
      }
    }
    __syncthreads();
  }
  for (int i = t; i < (TOPK + 31) / 32; i += 1024) keepOut[i] = kw[i];
}

// ---------------- finalize: stable compaction of kept boxes -> out [2000][5] ----------------
__global__ __launch_bounds__(1024) void k_final(const float* __restrict__ sB, const float* __restrict__ sE,
                                                const unsigned* __restrict__ kw, float* __restrict__ out) {
  __shared__ int cnt;
  if (threadIdx.x == 0) {
    int c = 0;
    for (int i = 0; i < TOPK && c < OUTK; ++i) {
      if (((kw[i >> 5] >> (i & 31)) & 1u) && sE[i] >= 0.f) {
        out[c * 5 + 0] = sB[i * 4 + 0]; out[c * 5 + 1] = sB[i * 4 + 1];
        out[c * 5 + 2] = sB[i * 4 + 2]; out[c * 5 + 3] = sB[i * 4 + 3];
        out[c * 5 + 4] = sE[i];
        ++c;
      }
    }
    cnt = c;
  }
  __syncthreads();
  for (int r = cnt + (int)threadIdx.x; r < OUTK; r += 1024) {
    out[r * 5 + 0] = 0.f; out[r * 5 + 1] = 0.f; out[r * 5 + 2] = 0.f;
    out[r * 5 + 3] = 0.f; out[r * 5 + 4] = 0.f;
  }
}

extern "C" void kernel_launch(void* const* d_in, const int* in_sizes, int n_in,
                              void* d_out, int out_size, void* d_ws, size_t ws_size,
                              hipStream_t stream) {
  const float* feat    = (const float*)d_in[1];
  const float* conv1_w = (const float*)d_in[2];
  const float* conv1_b = (const float*)d_in[3];
  const float* cls_w   = (const float*)d_in[4];
  const float* cls_b   = (const float*)d_in[5];
  const float* reg_w   = (const float*)d_in[6];
  const float* reg_b   = (const float*)d_in[7];
  float* out = (float*)d_out;

  char* w = (char*)d_ws;
  auto alloc = [&](size_t bytes) { void* p = (void*)w; w += (bytes + 255) & ~(size_t)255; return p; };
  _Float16* A1   = (_Float16*)alloc((size_t)POS * K1 * 2);
  _Float16* B1   = (_Float16*)alloc((size_t)C * K1 * 2);
  _Float16* X    = (_Float16*)alloc((size_t)POS * C * 2);
  _Float16* B2   = (_Float16*)alloc((size_t)NH * C * 2);
  float*    bias2= (float*)alloc(NH * 4);
  float*    head = (float*)alloc((size_t)POS * NH * 4);
  float*    boxesAll = (float*)alloc((size_t)NANCH * 4 * 4);
  float*    effAll   = (float*)alloc((size_t)NANCH * 4);
  unsigned long long* keys = (unsigned long long*)alloc((size_t)SORTN * 8);
  float*    selB = (float*)alloc((size_t)SORT2N * 4 * 4);
  float*    selE = (float*)alloc((size_t)SORT2N * 4);
  float*    srtB = (float*)alloc((size_t)SORT2N * 4 * 4);
  float*    srtE = (float*)alloc((size_t)SORT2N * 4);
  unsigned* keepW = (unsigned*)alloc(((TOPK + 31) / 32) * 4);

  k_pack_b1<<<(C * K1 + 255) / 256, 256, 0, stream>>>(conv1_w, B1);
  k_pack_b2<<<(NH * C + 255) / 256, 256, 0, stream>>>(cls_w, reg_w, cls_b, reg_b, B2, bias2);
  k_im2col<<<(POS * K1 + 255) / 256, 256, 0, stream>>>(feat, A1);
  k_gemm1<<<dim3(POS / 128, C / 32), 128, 0, stream>>>(A1, B1, conv1_b, X);
  k_gemm2<<<POS / 64, 128, 0, stream>>>(X, B2, bias2, head);
  k_props<<<SORTN / 256, 256, 0, stream>>>(head, boxesAll, effAll, keys);
  for (int k = 2; k <= SORTN; k <<= 1)
    for (int j = k >> 1; j > 0; j >>= 1)
      k_bitonic<<<SORTN / 256, 256, 0, stream>>>(keys, k, j);
  k_gather<<<SORT2N / 256, 256, 0, stream>>>(keys, boxesAll, effAll, selB, selE);
  k_sort2<<<1, 1024, 0, stream>>>(selB, selE, srtB, srtE);
  k_nms<<<1, 1024, 0, stream>>>(srtB, keepW);
  k_final<<<1, 1024, 0, stream>>>(srtB, srtE, keepW, out);
}
